// LinearAttention_21595095564320
// MI455X (gfx1250) — matrix operand order repaired and checked by reading, not compiled, not run on hardware
//
#include <hip/hip_runtime.h>

// ---------------------------------------------------------------------------
// LinearAttention for MI455X (gfx1250): bf16 WMMA pipeline
//   x[16,4096,512] fp32 -> qkv GEMM -> softmax(k) -> context -> M -> final
// Matrix math via v_wmma_f32_16x16x32_bf16 (wave32); context GEMM stages its
// tiles through LDS with global_load_async_to_lds_b128 (ASYNCcnt) double
// buffering.
// ---------------------------------------------------------------------------

typedef __attribute__((ext_vector_type(16))) __bf16 v16bf;
typedef __attribute__((ext_vector_type(8)))  __bf16 v8bf;
typedef __attribute__((ext_vector_type(8)))  float  v8f;

// Workspace layout (bytes, 256-aligned)
static constexpr size_t OFF_WQKVT = 0;                                   // 1536*512 bf16
static constexpr size_t OFF_WOUTT = OFF_WQKVT + (size_t)1536*512*2;      // 512*512 bf16
static constexpr size_t OFF_Q     = OFF_WOUTT + (size_t)512*512*2;       // [b,n,512] bf16
static constexpr size_t OFF_K     = OFF_Q  + (size_t)65536*512*2;        // [b,h,c,n] bf16
static constexpr size_t OFF_V     = OFF_K  + (size_t)65536*512*2;        // [b,h,c,n] bf16
static constexpr size_t OFF_CTX   = OFF_V  + (size_t)65536*512*2;        // [b,h,64,64] bf16
static constexpr size_t OFF_MT    = OFF_CTX + (size_t)128*64*64*2;       // [b,f,512] bf16

// ----------------------------- fragment loaders ----------------------------
// A (16x32 bf16, row-major src): lanes 0-15 row M=lane hold K {0..7,16..23},
// lanes 16-31 same rows hold K {8..15,24..31}; v16bf elems 0..7 -> VGPR0..3.
__device__ __forceinline__ v16bf load_a_rm_bf16(const __bf16* tile, int ld, int lane) {
  const int r    = lane & 15;
  const int koff = (lane & 16) ? 8 : 0;
  const __bf16* p = tile + (size_t)r * ld + koff;
  v8bf lo = *(const v8bf*)(p);
  v8bf hi = *(const v8bf*)(p + 16);
  return __builtin_shufflevector(lo, hi, 0,1,2,3,4,5,6,7,8,9,10,11,12,13,14,15);
}

// Same A fragment but sourced from fp32 (convert in-register)
__device__ __forceinline__ v16bf load_a_rm_f32(const float* tile, int ld, int lane) {
  const int r    = lane & 15;
  const int koff = (lane & 16) ? 8 : 0;
  const float* p = tile + (size_t)r * ld + koff;
  v8f lo = *(const v8f*)(p);
  v8f hi = *(const v8f*)(p + 16);
  v16bf a;
#pragma unroll
  for (int i = 0; i < 8; ++i) { a[i] = (__bf16)lo[i]; a[i + 8] = (__bf16)hi[i]; }
  return a;
}

// B (32x16 bf16) from column-major storage (each column's K contiguous):
// lanes 0-15: col N=lane, K {0..7,16..23}; lanes 16-31: col N=lane-16, K {8..15,24..31}.
__device__ __forceinline__ v16bf load_b_cm_bf16(const __bf16* tile, int ld, int lane) {
  const int c    = lane & 15;
  const int koff = (lane & 16) ? 8 : 0;
  const __bf16* p = tile + (size_t)c * ld + koff;
  v8bf lo = *(const v8bf*)(p);
  v8bf hi = *(const v8bf*)(p + 16);
  return __builtin_shufflevector(lo, hi, 0,1,2,3,4,5,6,7,8,9,10,11,12,13,14,15);
}

__device__ __forceinline__ v8f wmma_bf16(v16bf a, v16bf b, v8f c) {
  return __builtin_amdgcn_wmma_f32_16x16x32_bf16(false, a, false, b, (short)0, c,
                                                 false, false);
}

// Async 16-byte global->LDS copy (per lane); tracked by ASYNCcnt.
__device__ __forceinline__ void async_copy_b128(unsigned lds_addr, const void* gaddr) {
  asm volatile("global_load_async_to_lds_b128 %0, %1, off"
               :: "v"(lds_addr), "v"((unsigned long long)(uintptr_t)gaddr)
               : "memory");
}

// ----------------------- weight transpose+convert --------------------------
__global__ __launch_bounds__(256) void la_cvt_wqkv(const float* __restrict__ w,
                                                   __bf16* __restrict__ wT) {
  int idx = blockIdx.x * 256 + threadIdx.x;          // 512*1536
  int kk = idx / 1536, j = idx % 1536;
  wT[(size_t)j * 512 + kk] = (__bf16)w[idx];
}

__global__ __launch_bounds__(256) void la_cvt_wout(const float* __restrict__ w,
                                                   __bf16* __restrict__ wT) {
  int idx = blockIdx.x * 256 + threadIdx.x;          // 512*512
  int kk = idx >> 9, f = idx & 511;
  wT[(size_t)f * 512 + kk] = (__bf16)w[idx];
}

// ------------------------------- qkv GEMM ----------------------------------
// C[65536,1536] = x(f32->bf16) @ w_qkvT ; scatter epilogue into q/k/v layouts.
// grid(512, 24), block 256 (8 waves in 4x2), block tile 128(M) x 64(N).
__global__ __launch_bounds__(256) void la_gemm_qkv(const float* __restrict__ x,
                                                   const __bf16* __restrict__ wT,
                                                   __bf16* __restrict__ qb,
                                                   __bf16* __restrict__ kb,
                                                   __bf16* __restrict__ vb) {
  const int lane = threadIdx.x & 31;
  const int wave = threadIdx.x >> 5;
  const int wm = wave >> 1, wn = wave & 1;
  const int m0 = blockIdx.x * 128 + wm * 32;
  const int j0 = blockIdx.y * 64;
  const int n0 = j0 + wn * 32;

  v8f acc00 = {}, acc01 = {}, acc10 = {}, acc11 = {};
  for (int k0 = 0; k0 < 512; k0 += 32) {
    __builtin_prefetch(x + (size_t)m0 * 512 + k0 + 64, 0, 0);  // global_prefetch_b8
    v16bf a0 = load_a_rm_f32(x + (size_t)m0 * 512 + k0, 512, lane);
    v16bf a1 = load_a_rm_f32(x + (size_t)(m0 + 16) * 512 + k0, 512, lane);
    v16bf b0 = load_b_cm_bf16(wT + (size_t)n0 * 512 + k0, 512, lane);
    v16bf b1 = load_b_cm_bf16(wT + (size_t)(n0 + 16) * 512 + k0, 512, lane);
    acc00 = wmma_bf16(a0, b0, acc00);
    acc01 = wmma_bf16(a0, b1, acc01);
    acc10 = wmma_bf16(a1, b0, acc10);
    acc11 = wmma_bf16(a1, b1, acc11);
  }

  // Uniform per block: which matrix (q/k/v), head h, batch b.
  const int which = j0 >> 9;
  const int h     = (j0 >> 6) & 7;
  const int b     = m0 >> 12;             // 128-row tiles never cross a batch
  v8f accs[2][2] = {{acc00, acc01}, {acc10, acc11}};

  if (which == 0) {
#pragma unroll
    for (int tm = 0; tm < 2; ++tm)
#pragma unroll
      for (int tn = 0; tn < 2; ++tn)
#pragma unroll
        for (int i = 0; i < 8; ++i) {
          const int m  = m0 + tm * 16 + i + ((lane & 16) ? 8 : 0);
          const int hd = (n0 + tn * 16 + (lane & 15)) & 511;
          qb[(size_t)m * 512 + hd] = (__bf16)accs[tm][tn][i];   // [b,n,hd]
        }
  } else {
    __bf16* dst = (which == 1) ? kb : vb;
    const size_t bh64 = ((size_t)((b << 3) + h)) * 64;
#pragma unroll
    for (int tm = 0; tm < 2; ++tm)
#pragma unroll
      for (int tn = 0; tn < 2; ++tn)
#pragma unroll
        for (int i = 0; i < 8; ++i) {
          const int n = (m0 + tm * 16 + i + ((lane & 16) ? 8 : 0)) & 4095;
          const int c = (n0 + tn * 16 + (lane & 15)) & 63;
          dst[(bh64 + c) * 4096 + n] = (__bf16)accs[tm][tn][i]; // [b,h,c,n]
        }
  }
}

// ------------------------- softmax over n (in place) -----------------------
// One block per (b,h,c) row of k: 4096 bf16 elements, 256 threads x 16.
__global__ __launch_bounds__(256) void la_softmax_k(__bf16* __restrict__ kb) {
  __shared__ float red[256];
  const int tid = threadIdx.x;
  __bf16* row = kb + (size_t)blockIdx.x * 4096;

  float v[16];
#pragma unroll
  for (int i = 0; i < 16; ++i) v[i] = (float)row[tid + i * 256];

  float m = v[0];
#pragma unroll
  for (int i = 1; i < 16; ++i) m = fmaxf(m, v[i]);
  red[tid] = m; __syncthreads();
  for (int s = 128; s > 0; s >>= 1) {
    if (tid < s) red[tid] = fmaxf(red[tid], red[tid + s]);
    __syncthreads();
  }
  m = red[0]; __syncthreads();

  float sum = 0.f;
#pragma unroll
  for (int i = 0; i < 16; ++i) { v[i] = __expf(v[i] - m); sum += v[i]; }
  red[tid] = sum; __syncthreads();
  for (int s = 128; s > 0; s >>= 1) {
    if (tid < s) red[tid] += red[tid + s];
    __syncthreads();
  }
  const float inv = 1.f / red[0];
#pragma unroll
  for (int i = 0; i < 16; ++i) row[tid + i * 256] = (__bf16)(v[i] * inv);
}

// ----------------------------- context GEMM --------------------------------
// Per (b,h): ctx[64,64] = k_norm[64,4096] @ v[4096,64]. 128 blocks of 512
// threads (16 waves, one 16x16 tile each). K tiles (64x32 A + 64x32 B = 8 KB)
// are staged in LDS via async global->LDS copies, double buffered: each of
// the 512 threads issues one global_load_async_to_lds_b128 per K-step.
__global__ __launch_bounds__(512) void la_gemm_ctx(const __bf16* __restrict__ kb,
                                                   const __bf16* __restrict__ vb,
                                                   __bf16* __restrict__ ctx) {
  __shared__ __align__(32) __bf16 lds[2][2][64][32];   // [buf][A/B][row][k] 16 KB

  const int bh  = blockIdx.x;
  const int tid = threadIdx.x;
  const int lane = tid & 31;
  const int wave = tid >> 5;
  const int mt = wave >> 2, nt = wave & 3;

  const __bf16* abase = kb + (size_t)bh * 64 * 4096;   // rows d, K contiguous
  const __bf16* bbase = vb + (size_t)bh * 64 * 4096;   // cols e, K contiguous

  // Per-thread slice of the cooperative copy: 512 threads x 16 B = 8 KB.
  const int half = tid >> 8;              // 0 = A tile, 1 = B tile
  const int o    = (tid & 255) * 16;      // byte offset within 4 KB tile
  const int row  = o >> 6;                // 64 B per row (32 x bf16)
  const int kel  = (o & 63) >> 1;         // element offset within row
  const __bf16* gsrc = (half ? bbase : abase) + (size_t)row * 4096 + kel;
  const unsigned lds0 = (unsigned)(uintptr_t)(&lds[0][0][0][0]);
  const unsigned ldst = lds0 + (unsigned)(half * 4096 + o);

  // Prologue: fill buffer 0 with K-step 0.
  async_copy_b128(ldst, gsrc);

  v8f acc = {};
  for (int it = 0; it < 128; ++it) {
    if (it + 1 < 128) {
      // Kick off next tile into the other buffer, then wait for current.
      async_copy_b128(ldst + ((unsigned)(it + 1) & 1) * 8192,
                      gsrc + (size_t)(it + 1) * 32);
      asm volatile("s_wait_asynccnt 0x1" ::: "memory");
    } else {
      asm volatile("s_wait_asynccnt 0x0" ::: "memory");
    }
    __syncthreads();                       // all waves' copies of buf visible

    const int buf = it & 1;
    v16bf a = load_a_rm_bf16(&lds[buf][0][mt * 16][0], 32, lane);
    v16bf b = load_b_cm_bf16(&lds[buf][1][nt * 16][0], 32, lane);
    acc = wmma_bf16(a, b, acc);

    __syncthreads();                       // everyone done reading buf before
  }                                        // it gets overwritten next round

#pragma unroll
  for (int i = 0; i < 8; ++i) {
    const int d = mt * 16 + i + ((lane & 16) ? 8 : 0);
    const int e = nt * 16 + (lane & 15);
    ctx[((size_t)bh * 64 + d) * 64 + e] = (__bf16)acc[i];
  }
}

// ------------------------------- M GEMM ------------------------------------
// Per (b,h): M[64,512] = ctx[64,64] @ w_out[h*64: h*64+64, :]; store M^T
// as MT[b, f, hd] (col-major B for the final GEMM). 128 blocks, 16 waves.
__global__ __launch_bounds__(512) void la_gemm_m(const __bf16* __restrict__ ctx,
                                                 const __bf16* __restrict__ woT,
                                                 __bf16* __restrict__ MT) {
  const int bh = blockIdx.x;
  const int b = bh >> 3, h = bh & 7;
  const int lane = threadIdx.x & 31;
  const int wave = threadIdx.x >> 5;
  const int mt = wave & 3;          // d tile
  const int nt8 = wave >> 2;        // f block of 128
  const __bf16* abase = ctx + ((size_t)bh * 64 + mt * 16) * 64;

  v8f acc[8] = {};
  for (int k0 = 0; k0 < 64; k0 += 32) {
    v16bf a = load_a_rm_bf16(abase + k0, 64, lane);
#pragma unroll
    for (int t = 0; t < 8; ++t) {
      const __bf16* bb = woT + (size_t)(nt8 * 128 + t * 16) * 512 + h * 64 + k0;
      v16bf bf = load_b_cm_bf16(bb, 512, lane);
      acc[t] = wmma_bf16(a, bf, acc[t]);
    }
  }
#pragma unroll
  for (int t = 0; t < 8; ++t)
#pragma unroll
    for (int i = 0; i < 8; ++i) {
      const int f = nt8 * 128 + t * 16 + (lane & 15);
      const int d = mt * 16 + i + ((lane & 16) ? 8 : 0);
      MT[((size_t)b * 512 + f) * 512 + h * 64 + d] = (__bf16)acc[t][i];
    }
}

// ------------------------------ final GEMM ---------------------------------
// Per b: out_pre[4096,512] = Q[b,n,hd] @ M_b ; + bias, spatial (x,y) swap on
// store, fp32 output. grid(32, 8, 16), block 256 (8 waves 4x2), tile 128x64.
__global__ __launch_bounds__(256) void la_gemm_out(const __bf16* __restrict__ qb,
                                                   const __bf16* __restrict__ MT,
                                                   const float* __restrict__ bout,
                                                   float* __restrict__ out) {
  const int b = blockIdx.z;
  const int lane = threadIdx.x & 31;
  const int wave = threadIdx.x >> 5;
  const int wm = wave >> 1, wn = wave & 1;
  const int m0 = blockIdx.x * 128 + wm * 32;   // n within image
  const int f0 = blockIdx.y * 64 + wn * 32;    // output channel
  const __bf16* A = qb + (size_t)b * 4096 * 512;
  const __bf16* B = MT + (size_t)b * 512 * 512;

  v8f acc00 = {}, acc01 = {}, acc10 = {}, acc11 = {};
  for (int k0 = 0; k0 < 512; k0 += 32) {
    v16bf a0 = load_a_rm_bf16(A + (size_t)m0 * 512 + k0, 512, lane);
    v16bf a1 = load_a_rm_bf16(A + (size_t)(m0 + 16) * 512 + k0, 512, lane);
    v16bf b0 = load_b_cm_bf16(B + (size_t)f0 * 512 + k0, 512, lane);
    v16bf b1 = load_b_cm_bf16(B + (size_t)(f0 + 16) * 512 + k0, 512, lane);
    acc00 = wmma_bf16(a0, b0, acc00);
    acc01 = wmma_bf16(a0, b1, acc01);
    acc10 = wmma_bf16(a1, b0, acc10);
    acc11 = wmma_bf16(a1, b1, acc11);
  }

  v8f accs[2][2] = {{acc00, acc01}, {acc10, acc11}};
#pragma unroll
  for (int tm = 0; tm < 2; ++tm)
#pragma unroll
    for (int tn = 0; tn < 2; ++tn)
#pragma unroll
      for (int i = 0; i < 8; ++i) {
        const int n = m0 + tm * 16 + i + ((lane & 16) ? 8 : 0);
        const int f = f0 + tn * 16 + (lane & 15);
        const int xs = n >> 6, ys = n & 63;  // n = x*W + y ; out row = y*H + x
        out[(((size_t)b * 64 + ys) * 64 + xs) * 512 + f] = accs[tm][tn][i] + bout[f];
      }
}

// ------------------------------- launcher ----------------------------------
extern "C" void kernel_launch(void* const* d_in, const int* in_sizes, int n_in,
                              void* d_out, int out_size, void* d_ws, size_t ws_size,
                              hipStream_t stream) {
  (void)in_sizes; (void)n_in; (void)out_size; (void)ws_size;
  const float* x     = (const float*)d_in[0];
  const float* wqkv  = (const float*)d_in[1];
  const float* wout  = (const float*)d_in[2];
  const float* bout  = (const float*)d_in[3];
  float* out = (float*)d_out;

  char* ws = (char*)d_ws;
  __bf16* wqkvT = (__bf16*)(ws + OFF_WQKVT);
  __bf16* woutT = (__bf16*)(ws + OFF_WOUTT);
  __bf16* qb    = (__bf16*)(ws + OFF_Q);
  __bf16* kb    = (__bf16*)(ws + OFF_K);
  __bf16* vb    = (__bf16*)(ws + OFF_V);
  __bf16* ctx   = (__bf16*)(ws + OFF_CTX);
  __bf16* MT    = (__bf16*)(ws + OFF_MT);

  la_cvt_wqkv<<<(512 * 1536) / 256, 256, 0, stream>>>(wqkv, wqkvT);
  la_cvt_wout<<<(512 * 512) / 256, 256, 0, stream>>>(wout, woutT);
  la_gemm_qkv<<<dim3(512, 24), 256, 0, stream>>>(x, wqkvT, qb, kb, vb);
  la_softmax_k<<<8192, 256, 0, stream>>>(kb);
  la_gemm_ctx<<<128, 512, 0, stream>>>(kb, vb, ctx);
  la_gemm_m<<<128, 512, 0, stream>>>(ctx, woutT, MT);
  la_gemm_out<<<dim3(32, 8, 16), 256, 0, stream>>>(qb, MT, bout, out);
}
